// ChunkParallelProcessor_86887188398376
// MI455X (gfx1250) — matrix operand order repaired and checked by reading, not compiled, not run on hardware
//
#include <hip/hip_runtime.h>
#include <hip/hip_bf16.h>

// ---------------------------------------------------------------------------
// GRU (chunk-parallel output layout) for MI455X / gfx1250, wave32 + WMMA bf16.
//
//   z = sigmoid([x,h]Wz+bz); r = sigmoid([x,h]Wr+br)
//   hc = tanh([x, r*h]Wh+bh); h' = (1-z)h + z*hc
//
// Pipeline:
//   (1) convert x -> bf16 (once)
//   (2) swizzle Wzr / Wh -> WMMA B-operand tile layout in bf16 (once)
//   (3) init h buffers from h0
//   (4) per t: S1 kernel (z/r gates + rh), S2 kernel (candidate + update + store)
// Each wave (32 threads) owns a 16x32 output slab (two 16x16 tiles, shared A);
// K-loop is split into an x-phase and a recurrent-phase (uniform addressing,
// no per-iteration selects); 128 v_wmma_f32_16x16x32_bf16 per tile pair.
// Epilogue activations use hardware transcendentals (v_exp_f32 / v_rcp_f32).
// ---------------------------------------------------------------------------

typedef __attribute__((ext_vector_type(16))) __bf16 v16bf;
typedef __attribute__((ext_vector_type(8)))  __bf16 v8bf;
typedef __attribute__((ext_vector_type(8)))  float  v8f;

static constexpr int kB = 64;
static constexpr int kT = 512;
static constexpr int kD = 1024;
static constexpr int kH = 1024;
static constexpr int kK = kD + kH;          // 2048, GEMM reduction dim
static constexpr int kKT = kK / 32;         // 64 K-chunks per tile column
static constexpr size_t kOutsElems = (size_t)kB * kT * kH;   // finals follow

// round-to-nearest-even fp32 -> bf16 bits
__device__ __forceinline__ unsigned short f2bf(float f) {
    unsigned int u = __float_as_uint(f);
    u += 0x7FFFu + ((u >> 16) & 1u);
    return (unsigned short)(u >> 16);
}

__device__ __forceinline__ v16bf cat16(v8bf lo, v8bf hi) {
    return __builtin_shufflevector(lo, hi, 0,1,2,3,4,5,6,7,8,9,10,11,12,13,14,15);
}

// hardware-transcendental activations: v_exp_f32 + v_rcp_f32
__device__ __forceinline__ float fast_sigmoid(float x) {
    return __builtin_amdgcn_rcpf(1.0f + __expf(-x));
}
__device__ __forceinline__ float fast_tanh(float x) {
    return __builtin_fmaf(2.0f, fast_sigmoid(2.0f * x), -1.0f);
}

// ---------------------------------------------------------------------------
// (1) x fp32 -> bf16
__global__ void cvt_x_bf16(const float* __restrict__ x, unsigned short* __restrict__ xb, size_t n) {
    size_t i = (size_t)blockIdx.x * blockDim.x + threadIdx.x;
    size_t stride = (size_t)gridDim.x * blockDim.x;
    for (; i < n; i += stride) xb[i] = f2bf(x[i]);
}

// ---------------------------------------------------------------------------
// (2) weight swizzle into WMMA B-operand layout.
// Tile (kt, nt) covers K rows [kt*32, kt*32+32), N cols [nt*16, nt*16+16).
// Storage: flat[(nt*kKT + kt)*512 + lane*16 + e]
//   lane < 16 : n = nt*16+lane,    k = kt*32 + {0..7, 16..23}   (e = 0..15)
//   lane >= 16: n = nt*16+lane-16, k = kt*32 + {8..15, 24..31}
__global__ void swizzle_wzr(const float* __restrict__ Wz, const float* __restrict__ Wr,
                            unsigned short* __restrict__ out) {
    size_t i = (size_t)blockIdx.x * blockDim.x + threadIdx.x;     // < 2048*2048
    size_t tile = i >> 9;
    int within = (int)(i & 511);
    int lane = within >> 4, e = within & 15;
    int kt = (int)(tile & (kKT - 1));
    int nt = (int)(tile >> 6);
    int ng = nt * 16 + (lane & 15);
    int k  = kt * 32 + ((lane >= 16) ? 8 : 0) + e + ((e >= 8) ? 8 : 0);
    const float* W = (ng < kH) ? Wz : Wr;
    int n = ng & (kH - 1);
    out[i] = f2bf(W[(size_t)k * kH + n]);
}

__global__ void swizzle_wh(const float* __restrict__ Wh, unsigned short* __restrict__ out) {
    size_t i = (size_t)blockIdx.x * blockDim.x + threadIdx.x;     // < 2048*1024
    size_t tile = i >> 9;
    int within = (int)(i & 511);
    int lane = within >> 4, e = within & 15;
    int kt = (int)(tile & (kKT - 1));
    int nt = (int)(tile >> 6);
    int n = nt * 16 + (lane & 15);
    int k = kt * 32 + ((lane >= 16) ? 8 : 0) + e + ((e >= 8) ? 8 : 0);
    out[i] = f2bf(Wh[(size_t)k * kH + n]);
}

// ---------------------------------------------------------------------------
// (3) h0 -> h_f32, h_bf16
__global__ void init_h(const float* __restrict__ h0, float* __restrict__ hf,
                       unsigned short* __restrict__ hb, int n) {
    int i = blockIdx.x * blockDim.x + threadIdx.x;
    if (i < n) { float v = h0[i]; hf[i] = v; hb[i] = f2bf(v); }
}

// ---------------------------------------------------------------------------
// A fragment from a row-major bf16 matrix (row stride given by caller's base):
//   lane < 16 : k = k0 + {0..7, 16..23};  lane >= 16: k = k0 + {8..15, 24..31}
__device__ __forceinline__ v16bf load_a(const unsigned short* __restrict__ base) {
    v8bf lo = *(const v8bf*)(base);
    v8bf hi = *(const v8bf*)(base + 16);
    return cat16(lo, hi);
}
__device__ __forceinline__ v16bf load_b(const unsigned short* __restrict__ p) {
    v8bf lo = *(const v8bf*)(p);
    v8bf hi = *(const v8bf*)(p + 8);
    return cat16(lo, hi);
}

// Shared GEMM core: two adjacent 16x16 N-tiles per wave, A reused.
//   xa : &x_bf16[row, t, 0]   (x-phase,   kt = 0..31)
//   ra : &rec_bf16[row, 0]    (rec-phase, kt = 32..63)
//   wp : B tiles for nt0, wq : B tiles for nt0+1 (each kKT consecutive tiles)
__device__ __forceinline__ void gemm_pair(const unsigned short* __restrict__ xa,
                                          const unsigned short* __restrict__ ra,
                                          const unsigned short* __restrict__ wp,
                                          const unsigned short* __restrict__ wq,
                                          v8f& acc0, v8f& acc1) {
    #pragma unroll 4
    for (int ki = 0; ki < kKT / 2; ++ki) {          // x-phase
        v16bf A  = load_a(xa + ki * 32);
        v16bf B0 = load_b(wp);
        v16bf B1 = load_b(wq);
        __builtin_prefetch(wp + 1024, 0, 1);
        __builtin_prefetch(wq + 1024, 0, 1);
        acc0 = __builtin_amdgcn_wmma_f32_16x16x32_bf16(false, A, false, B0,
                                                       (short)0, acc0, false, false);
        acc1 = __builtin_amdgcn_wmma_f32_16x16x32_bf16(false, A, false, B1,
                                                       (short)0, acc1, false, false);
        wp += 512; wq += 512;
    }
    #pragma unroll 4
    for (int ki = 0; ki < kKT / 2; ++ki) {          // recurrent phase
        v16bf A  = load_a(ra + ki * 32);
        v16bf B0 = load_b(wp);
        v16bf B1 = load_b(wq);
        __builtin_prefetch(wp + 1024, 0, 1);
        __builtin_prefetch(wq + 1024, 0, 1);
        acc0 = __builtin_amdgcn_wmma_f32_16x16x32_bf16(false, A, false, B0,
                                                       (short)0, acc0, false, false);
        acc1 = __builtin_amdgcn_wmma_f32_16x16x32_bf16(false, A, false, B1,
                                                       (short)0, acc1, false, false);
        wp += 512; wq += 512;
    }
}

// ---------------------------------------------------------------------------
// (4a) S1: G = [x_t,h] @ [Wz|Wr];  z = sigmoid(G_z+bz) -> z_ws
//                                  r = sigmoid(G_r+br); rh = bf16(r*h) -> rh_ws
__global__ __launch_bounds__(32)
void gru_step_zr(int t,
                 const unsigned short* __restrict__ xb,
                 const unsigned short* __restrict__ hb,
                 const float* __restrict__ hf,
                 const unsigned short* __restrict__ Wzr,
                 const float* __restrict__ bz, const float* __restrict__ br,
                 float* __restrict__ z_ws, unsigned short* __restrict__ rh_ws) {
    const int lane = threadIdx.x;
    const int nt0 = blockIdx.x * 2;         // 0..126 step 2  (N = 2048)
    const int mt  = blockIdx.y;             // 0..3           (M = 64)
    const int lr = lane & 15;
    const int hiHalf = lane >> 4;
    const int row = mt * 16 + lr;
    const int aoff = hiHalf ? 8 : 0;

    const unsigned short* xa = xb + ((size_t)row * kT + t) * kD + aoff;
    const unsigned short* ra = hb + (size_t)row * kH + aoff;
    const unsigned short* wp = Wzr + ((size_t)nt0       * kKT) * 512 + lane * 16;
    const unsigned short* wq = Wzr + ((size_t)(nt0 + 1) * kKT) * 512 + lane * 16;

    v8f acc0 = {}, acc1 = {};
    gemm_pair(xa, ra, wp, wq, acc0, acc1);

    const int n0 = nt0 * 16 + lr;           // tile pair never straddles z/r split
    if (n0 < kH) {                          // z gate
        const float bz0 = bz[n0], bz1 = bz[n0 + 16];
        #pragma unroll
        for (int j = 0; j < 8; ++j) {
            int m = mt * 16 + j + hiHalf * 8;
            z_ws[m * kH + n0]      = fast_sigmoid(acc0[j] + bz0);
            z_ws[m * kH + n0 + 16] = fast_sigmoid(acc1[j] + bz1);
        }
    } else {                                // r gate -> r*h
        const int n2 = n0 - kH;
        const float br0 = br[n2], br1 = br[n2 + 16];
        #pragma unroll
        for (int j = 0; j < 8; ++j) {
            int m = mt * 16 + j + hiHalf * 8;
            float r0 = fast_sigmoid(acc0[j] + br0);
            float r1 = fast_sigmoid(acc1[j] + br1);
            rh_ws[m * kH + n2]      = f2bf(r0 * hf[m * kH + n2]);
            rh_ws[m * kH + n2 + 16] = f2bf(r1 * hf[m * kH + n2 + 16]);
        }
    }
}

// ---------------------------------------------------------------------------
// (4b) S2: hc = tanh([x_t, rh] @ Wh + bh);  h' = (1-z)h + z*hc
//      store outs[b,t,:], finals on t%4==3, refresh h_f32 / h_bf16.
__global__ __launch_bounds__(32)
void gru_step_h(int t,
                const unsigned short* __restrict__ xb,
                const unsigned short* __restrict__ rh_ws,
                const unsigned short* __restrict__ Wh,
                const float* __restrict__ bh,
                const float* __restrict__ z_ws,
                float* __restrict__ hf, unsigned short* __restrict__ hb,
                float* __restrict__ out) {
    const int lane = threadIdx.x;
    const int nt0 = blockIdx.x * 2;         // 0..62 step 2   (N = 1024)
    const int mt  = blockIdx.y;             // 0..3
    const int lr = lane & 15;
    const int hiHalf = lane >> 4;
    const int row = mt * 16 + lr;
    const int aoff = hiHalf ? 8 : 0;

    const unsigned short* xa = xb + ((size_t)row * kT + t) * kD + aoff;
    const unsigned short* ra = rh_ws + (size_t)row * kH + aoff;
    const unsigned short* wp = Wh + ((size_t)nt0       * kKT) * 512 + lane * 16;
    const unsigned short* wq = Wh + ((size_t)(nt0 + 1) * kKT) * 512 + lane * 16;

    v8f acc0 = {}, acc1 = {};
    gemm_pair(xa, ra, wp, wq, acc0, acc1);

    const int n0 = nt0 * 16 + lr;
    const float bh0 = bh[n0], bh1 = bh[n0 + 16];
    #pragma unroll
    for (int j = 0; j < 8; ++j) {
        int m = mt * 16 + j + hiHalf * 8;   // batch row
        #pragma unroll
        for (int s = 0; s < 2; ++s) {
            int  n  = n0 + s * 16;
            float g = (s ? acc1[j] + bh1 : acc0[j] + bh0);
            float hc = fast_tanh(g);
            float z  = z_ws[m * kH + n];
            float hp = hf[m * kH + n];
            float hn = __builtin_fmaf(z, hc - hp, hp);   // (1-z)h + z*hc
            out[((size_t)m * kT + t) * kH + n] = hn;                       // outs
            if ((t & 3) == 3)
                out[kOutsElems + ((size_t)m * (kT / 4) + (t >> 2)) * kH + n] = hn;
            hf[m * kH + n] = hn;
            hb[m * kH + n] = f2bf(hn);
        }
    }
}

// ---------------------------------------------------------------------------
extern "C" void kernel_launch(void* const* d_in, const int* in_sizes, int n_in,
                              void* d_out, int out_size, void* d_ws, size_t ws_size,
                              hipStream_t stream) {
    const float* x  = (const float*)d_in[0];
    const float* h0 = (const float*)d_in[1];
    const float* Wz = (const float*)d_in[2];
    const float* bz = (const float*)d_in[3];
    const float* Wr = (const float*)d_in[4];
    const float* br = (const float*)d_in[5];
    const float* Wh = (const float*)d_in[6];
    const float* bh = (const float*)d_in[7];
    float* out = (float*)d_out;

    // workspace carve-up (all 256B aligned)
    char* ws = (char*)d_ws;
    size_t off = 0;
    auto take = [&](size_t bytes) { char* p = ws + off; off += (bytes + 255) & ~(size_t)255; return p; };
    unsigned short* xb     = (unsigned short*)take((size_t)kB * kT * kD * 2);  // 64 MB
    unsigned short* Wzr_sw = (unsigned short*)take((size_t)kK * 2 * kH * 2);   // 8 MB
    unsigned short* Wh_sw  = (unsigned short*)take((size_t)kK * kH * 2);       // 4 MB
    float*          hf     = (float*)take((size_t)kB * kH * 4);
    unsigned short* hb     = (unsigned short*)take((size_t)kB * kH * 2);
    float*          z_ws   = (float*)take((size_t)kB * kH * 4);
    unsigned short* rh_ws  = (unsigned short*)take((size_t)kB * kH * 2);
    (void)ws_size; (void)in_sizes; (void)n_in; (void)out_size;

    // (1) x -> bf16
    cvt_x_bf16<<<4096, 256, 0, stream>>>(x, xb, (size_t)kB * kT * kD);
    // (2) weight swizzles
    swizzle_wzr<<<(int)(((size_t)kK * 2 * kH) / 256), 256, 0, stream>>>(Wz, Wr, Wzr_sw);
    swizzle_wh<<<(int)(((size_t)kK * kH) / 256), 256, 0, stream>>>(Wh, Wh_sw);
    // (3) h init
    init_h<<<(kB * kH) / 256, 256, 0, stream>>>(h0, hf, hb, kB * kH);

    // (4) recurrence: 512 dependent step pairs
    for (int t = 0; t < kT; ++t) {
        gru_step_zr<<<dim3(kH / 16, kB / 16), 32, 0, stream>>>(      // 64 x 4 blocks
            t, xb, hb, hf, Wzr_sw, bz, br, z_ws, rh_ws);
        gru_step_h<<<dim3(kH / 32, kB / 16), 32, 0, stream>>>(       // 32 x 4 blocks
            t, xb, rh_ws, Wh_sw, bh, z_ws, hf, hb, out);
    }
}
